// VoiceSynthesizer_29987461660887
// MI455X (gfx1250) — matrix operand order repaired and checked by reading, not compiled, not run on hardware
//
#include <hip/hip_runtime.h>
#include <hip/hip_bf16.h>

#define B_ 8
#define S_ 128
#define T_ 256
#define MEL_ 80

typedef __bf16 v16bf __attribute__((ext_vector_type(16)));
typedef float v8f __attribute__((ext_vector_type(8)));
typedef unsigned int v4u __attribute__((ext_vector_type(4)));

union Frag {
  v16bf bf;
  v4u q[2];
};

__device__ __forceinline__ unsigned short f2bf(float f) {
  unsigned int u = __float_as_uint(f);
  u += 0x7FFFu + ((u >> 16) & 1u);
  return (unsigned short)(u >> 16);
}

__device__ __forceinline__ v8f wmma_bf16(const Frag& a, const Frag& b, v8f c) {
  return __builtin_amdgcn_wmma_f32_16x16x32_bf16(false, a.bf, false, b.bf,
                                                 (short)0, c, false, false);
}

__device__ __forceinline__ float sigf(float x) { return 1.0f / (1.0f + __expf(-x)); }

// ---------------------------------------------------------------------------
// Generic GEMM: C(MxN) = A(MxK)bf16 * Bw(Npad,K)bf16^T + bias1 + bias2 (f32).
// Bw is padded (zero rows) to Npad = multiple of 64, so the K-loop is fully
// branch-free (WMMA must run with EXEC all-1s). Stores are lane-guarded n<N.
// One wave computes a 16x64 C strip (4 WMMA tiles); K multiple of 32, M of 16.
// ---------------------------------------------------------------------------
__global__ __launch_bounds__(256) void gemm_bf16_kernel(
    const unsigned short* __restrict__ A, const unsigned short* __restrict__ Bw,
    float* __restrict__ C, const float* __restrict__ bias1,
    const float* __restrict__ bias2, int M, int N, int Npad, int K) {
  int wid = threadIdx.x >> 5;
  int lane = threadIdx.x & 31;
  int laneN = lane & 15;
  int hi = lane >> 4;
  int nMac = Npad >> 6;
  int total = (M >> 4) * nMac;
  int tile = blockIdx.x * 8 + wid;
  // scalar (wave-uniform) early exit: keep EXEC intact around all WMMAs
  if (__builtin_amdgcn_readfirstlane(tile) >= total) return;
  int mt = tile / nMac, nt = tile - mt * nMac;
  int mBase = mt << 4;
  v8f acc[4];
#pragma unroll
  for (int j = 0; j < 4; ++j)
#pragma unroll
    for (int r = 0; r < 8; ++r) acc[j][r] = 0.0f;
  const unsigned short* aRow = A + (size_t)(mBase + laneN) * K;
  const unsigned short* bRow0 =
      Bw + (size_t)((nt << 6) + laneN) * K + (hi ? 16 : 0);
  for (int kk = 0; kk < K; kk += 32) {
    Frag a;
    const v4u* pa = reinterpret_cast<const v4u*>(aRow + kk + hi * 16);
    a.q[0] = pa[0];
    a.q[1] = pa[1];
#pragma unroll
    for (int j = 0; j < 4; ++j) {
      const v4u* pb =
          reinterpret_cast<const v4u*>(bRow0 + (size_t)(j << 4) * K + kk);
      Frag b;
      b.q[0] = pb[0];
      b.q[1] = pb[1];
      acc[j] = wmma_bf16(a, b, acc[j]);
    }
  }
#pragma unroll
  for (int j = 0; j < 4; ++j) {
    int n = (nt << 6) + (j << 4) + laneN;
    if (n < N) {
      float bs = (bias1 ? bias1[n] : 0.0f) + (bias2 ? bias2[n] : 0.0f);
#pragma unroll
      for (int r = 0; r < 8; ++r)
        C[(size_t)(mBase + r + hi * 8) * N + n] = acc[j][r] + bs;
    }
  }
}

// ---------------------------------------------------------------------------
// Persistent LSTM scan. Gpre = x@Wih^T + bih + bhh precomputed, layout
// (T,16,4H). h kept as bf16 in LDS (16 x H row-major), c in accumulator regs.
// 32 waves; each wave owns NHT 16-wide hidden slices x 4 gates.
// blockIdx.x selects direction (0=fwd col=colF, 1=bwd reversed col=colB).
// ---------------------------------------------------------------------------
template <int H, int NHT>
__global__ __launch_bounds__(1024) void lstm_scan_kernel(
    const float* __restrict__ GpreF, const float* __restrict__ GpreB,
    const unsigned short* __restrict__ WhhF,
    const unsigned short* __restrict__ WhhB, float* __restrict__ hout,
    int out_ld, int colF, int colB, int T) {
  const float* Gpre = (blockIdx.x == 0) ? GpreF : GpreB;
  const unsigned short* Whh = (blockIdx.x == 0) ? WhhF : WhhB;
  int col = (blockIdx.x == 0) ? colF : colB;
  int rev = (blockIdx.x == 0) ? 0 : 1;
  __shared__ __attribute__((aligned(16))) unsigned short h_lds[16 * H];
  for (int i = threadIdx.x; i < 16 * H; i += blockDim.x) h_lds[i] = 0;
  __syncthreads();
  int wid = threadIdx.x >> 5, lane = threadIdx.x & 31;
  int laneN = lane & 15, hi = lane >> 4;
  v8f c_reg[NHT];
#pragma unroll
  for (int j = 0; j < NHT; ++j)
#pragma unroll
    for (int r = 0; r < 8; ++r) c_reg[j][r] = 0.0f;
  for (int t = 0; t < T; ++t) {
    int tt = rev ? (T - 1 - t) : t;
    const float* grow = Gpre + (size_t)tt * 16 * (4 * H);
    v8f acc[4][NHT];
#pragma unroll
    for (int g = 0; g < 4; ++g)
#pragma unroll
      for (int j = 0; j < NHT; ++j) {
        int n = g * H + (wid * NHT + j) * 16 + laneN;
#pragma unroll
        for (int r = 0; r < 8; ++r)
          acc[g][j][r] = grow[(size_t)(r + hi * 8) * (4 * H) + n];
      }
    for (int kk = 0; kk < H; kk += 32) {
      Frag a;
      const v4u* pa =
          reinterpret_cast<const v4u*>(&h_lds[laneN * H + kk + hi * 16]);
      a.q[0] = pa[0];
      a.q[1] = pa[1];
#pragma unroll
      for (int g = 0; g < 4; ++g)
#pragma unroll
        for (int j = 0; j < NHT; ++j) {
          int n = g * H + (wid * NHT + j) * 16 + laneN;
          const v4u* pb = reinterpret_cast<const v4u*>(Whh + (size_t)n * H +
                                                       kk + (hi ? 16 : 0));
          Frag b;
          b.q[0] = pb[0];
          b.q[1] = pb[1];
          acc[g][j] = wmma_bf16(a, b, acc[g][j]);
        }
    }
    __syncthreads();  // all h_lds reads for this step done
#pragma unroll
    for (int j = 0; j < NHT; ++j) {
      int hcol = (wid * NHT + j) * 16 + laneN;
#pragma unroll
      for (int r = 0; r < 8; ++r) {
        float iv = sigf(acc[0][j][r]);
        float fv = sigf(acc[1][j][r]);
        float gv = tanhf(acc[2][j][r]);
        float ov = sigf(acc[3][j][r]);
        float cv = fv * c_reg[j][r] + iv * gv;
        c_reg[j][r] = cv;
        float hv = ov * tanhf(cv);
        int brow = r + hi * 8;
        h_lds[brow * H + hcol] = f2bf(hv);
        hout[(size_t)(tt * 16 + brow) * out_ld + col + hcol] = hv;
      }
    }
    __syncthreads();  // new h visible before next step reads it
  }
}

// ---------------------------------------------------------------------------
// Cross attention: one block = (b, head, 64-query strip). hd=64, S=128.
// scores in LDS f32, P in LDS bf16, both matmuls via WMMA.
// ---------------------------------------------------------------------------
__global__ __launch_bounds__(256) void attention_kernel(
    const unsigned short* __restrict__ Qbf,
    const unsigned short* __restrict__ Kbf,
    const unsigned short* __restrict__ Vt, float* __restrict__ attnH) {
  int idx = blockIdx.x;
  int tq = idx & 3;
  int h8 = (idx >> 2) & 7;
  int b = idx >> 5;
  __shared__ __attribute__((aligned(16))) float s_sc[64 * 128];
  __shared__ __attribute__((aligned(16))) unsigned short s_p[64 * 128];
  int wid = threadIdx.x >> 5, lane = threadIdx.x & 31;
  int laneN = lane & 15, hi = lane >> 4;
  // phase 1: scores = (Q K^T) * 1/sqrt(64)
#pragma unroll
  for (int j = 0; j < 4; ++j) {
    int tid = wid * 4 + j;
    int mt = tid >> 3, nt = tid & 7;
    v8f acc;
#pragma unroll
    for (int r = 0; r < 8; ++r) acc[r] = 0.0f;
#pragma unroll
    for (int kk = 0; kk < 64; kk += 32) {
      Frag a, bb;
      const v4u* pa = reinterpret_cast<const v4u*>(
          Qbf + ((size_t)(tq * 64 + mt * 16 + laneN) * 16 + b) * 1024 +
          h8 * 64 + kk + hi * 16);
      a.q[0] = pa[0];
      a.q[1] = pa[1];
      const v4u* pb = reinterpret_cast<const v4u*>(
          Kbf + ((size_t)(nt * 16 + laneN) * 16 + b) * 1024 + h8 * 64 + kk +
          (hi ? 16 : 0));
      bb.q[0] = pb[0];
      bb.q[1] = pb[1];
      acc = wmma_bf16(a, bb, acc);
    }
#pragma unroll
    for (int r = 0; r < 8; ++r)
      s_sc[(mt * 16 + r + hi * 8) * 128 + nt * 16 + laneN] = acc[r] * 0.125f;
  }
  __syncthreads();
  if (threadIdx.x < 64) {  // row softmax
    float* row = s_sc + threadIdx.x * 128;
    float mx = row[0];
    for (int c = 1; c < 128; ++c) mx = fmaxf(mx, row[c]);
    float sum = 0.0f;
    for (int c = 0; c < 128; ++c) {
      float e = __expf(row[c] - mx);
      row[c] = e;
      sum += e;
    }
    float inv = 1.0f / sum;
    unsigned short* prow = s_p + threadIdx.x * 128;
    for (int c = 0; c < 128; ++c) prow[c] = f2bf(row[c] * inv);
  }
  __syncthreads();
  // phase 2: O = P V  (Vt is (b,h,64,128) so lanes read contiguous k)
#pragma unroll
  for (int j = 0; j < 2; ++j) {
    int tid = wid * 2 + j;
    int mt = tid >> 2, nt = tid & 3;
    v8f acc;
#pragma unroll
    for (int r = 0; r < 8; ++r) acc[r] = 0.0f;
#pragma unroll
    for (int kk = 0; kk < 128; kk += 32) {
      Frag a, bb;
      const v4u* pa =
          reinterpret_cast<const v4u*>(s_p + (mt * 16 + laneN) * 128 + kk + hi * 16);
      a.q[0] = pa[0];
      a.q[1] = pa[1];
      const v4u* pb = reinterpret_cast<const v4u*>(
          Vt + (size_t)((b * 8 + h8) * 64 + nt * 16 + laneN) * 128 + kk +
          (hi ? 16 : 0));
      bb.q[0] = pb[0];
      bb.q[1] = pb[1];
      acc = wmma_bf16(a, bb, acc);
    }
#pragma unroll
    for (int r = 0; r < 8; ++r)
      attnH[((size_t)(tq * 64 + mt * 16 + r + hi * 8) * 16 + b) * 1024 +
            h8 * 64 + nt * 16 + laneN] = acc[r];
  }
}

// ------------------------------ elementwise --------------------------------
__global__ void k_convert(const float* __restrict__ s,
                          unsigned short* __restrict__ d, size_t n) {
  size_t i = (size_t)blockIdx.x * blockDim.x + threadIdx.x;
  if (i < n) d[i] = f2bf(s[i]);
}
// (Nsrc,Ks) f32 -> (Ndst,Kd) bf16, zero-padded rows/cols
__global__ void k_convert_pad(const float* __restrict__ s,
                              unsigned short* __restrict__ d, int Nsrc,
                              int Ndst, int Ks, int Kd) {
  size_t i = (size_t)blockIdx.x * blockDim.x + threadIdx.x;
  size_t total = (size_t)Ndst * Kd;
  if (i >= total) return;
  int kc = (int)(i % Kd);
  size_t n = i / Kd;
  unsigned short v = 0;
  if ((int)n < Nsrc && kc < Ks) v = f2bf(s[n * Ks + kc]);
  d[i] = v;
}
// conv weight (O,I,5) f32 -> (Odst, Kd) bf16, patch order [k][c], zero-padded
__global__ void k_conv_repack(const float* __restrict__ W,
                              unsigned short* __restrict__ d, int O, int Odst,
                              int I, int Kd) {
  size_t i = (size_t)blockIdx.x * blockDim.x + threadIdx.x;
  size_t total = (size_t)Odst * Kd;
  if (i >= total) return;
  int kc = (int)(i % Kd);
  size_t o = i / Kd;
  unsigned short v = 0;
  if ((int)o < O && kc < 5 * I) {
    int k = kc / I;
    int c = kc - k * I;
    v = f2bf(W[((size_t)o * I + c) * 5 + k]);
  }
  d[i] = v;
}
__global__ void k_embed(const int* __restrict__ tok,
                        const float* __restrict__ emb,
                        unsigned short* __restrict__ d) {
  size_t i = (size_t)blockIdx.x * blockDim.x + threadIdx.x;
  size_t total = (size_t)S_ * 16 * 512;
  if (i >= total) return;
  int c = (int)(i % 512);
  size_t row = i / 512;
  int b = (int)(row % 16);
  int s = (int)(row / 16);
  unsigned short v = 0;
  if (b < 8) {
    int t = tok[b * S_ + s];
    v = f2bf(emb[(size_t)t * 512 + c]);
  }
  d[i] = v;
}
__global__ void k_spkA(const float* __restrict__ spk,
                       unsigned short* __restrict__ d) {
  int i = blockIdx.x * blockDim.x + threadIdx.x;
  if (i >= 16 * 256) return;
  int b = i / 256, c = i % 256;
  d[i] = (b < 8) ? f2bf(spk[b * 256 + c]) : (unsigned short)0;
}
__global__ void k_spk_add(const float* __restrict__ enc,
                          const float* __restrict__ spkP,
                          float* __restrict__ encS,
                          unsigned short* __restrict__ encSB) {
  size_t i = (size_t)blockIdx.x * blockDim.x + threadIdx.x;
  size_t total = (size_t)S_ * 16 * 1024;
  if (i >= total) return;
  int col = (int)(i % 1024);
  int b = (int)((i / 1024) % 16);
  float v = enc[i] + spkP[b * 1024 + col];
  encS[i] = v;
  encSB[i] = f2bf(v);
}
__global__ void k_qpad(const float* __restrict__ melT,
                       unsigned short* __restrict__ d) {
  size_t i = (size_t)blockIdx.x * blockDim.x + threadIdx.x;
  size_t total = (size_t)T_ * 16 * 1024;
  if (i >= total) return;
  int col = (int)(i % 1024);
  size_t row = i / 1024;
  int b = (int)(row % 16);
  int t = (int)(row / 16);
  unsigned short v = 0;
  if (b < 8 && col < MEL_ && t > 0)
    v = f2bf(melT[((size_t)b * T_ + (t - 1)) * MEL_ + col]);
  d[i] = v;
}
__global__ void k_buildX(const float* __restrict__ attn,
                         const float* __restrict__ melT,
                         unsigned short* __restrict__ d) {
  size_t i = (size_t)blockIdx.x * blockDim.x + threadIdx.x;
  size_t total = (size_t)T_ * 16 * 1120;
  if (i >= total) return;
  int col = (int)(i % 1120);
  size_t row = i / 1120;
  int b = (int)(row % 16);
  int t = (int)(row / 16);
  unsigned short v = 0;
  if (col < 1024) {
    if (b < 8) v = f2bf(attn[row * 1024 + col]);
  } else if (col < 1104) {
    int c = col - 1024;
    if (b < 8 && t > 0) v = f2bf(melT[((size_t)b * T_ + (t - 1)) * MEL_ + c]);
  }
  d[i] = v;
}
__global__ void k_vt(const float* __restrict__ Vf,
                     unsigned short* __restrict__ Vt) {
  size_t i = (size_t)blockIdx.x * blockDim.x + threadIdx.x;
  size_t total = (size_t)8 * 8 * 64 * 128;
  if (i >= total) return;
  int s = (int)(i % 128);
  size_t r = i / 128;
  int dch = (int)(r % 64);
  r /= 64;
  int h = (int)(r % 8);
  int b = (int)(r / 8);
  Vt[i] = f2bf(Vf[((size_t)s * 16 + b) * 1024 + h * 64 + dch]);
}
__global__ void k_bn_tanh(const float* __restrict__ src,
                          const float* __restrict__ g,
                          const float* __restrict__ be,
                          const float* __restrict__ m,
                          const float* __restrict__ vv, float* __restrict__ dst,
                          int C, size_t n) {
  size_t i = (size_t)blockIdx.x * blockDim.x + threadIdx.x;
  if (i >= n) return;
  int c = (int)(i % C);
  float x = tanhf(src[i]);
  dst[i] = (x - m[c]) * (g[c] * rsqrtf(vv[c] + 1e-5f)) + be[c];
}
__global__ void k_im2col(const float* __restrict__ src,
                         unsigned short* __restrict__ dst, int L, int C,
                         int Kd) {
  size_t i = (size_t)blockIdx.x * blockDim.x + threadIdx.x;
  size_t total = (size_t)L * 16 * Kd;
  if (i >= total) return;
  int kc = (int)(i % Kd);
  size_t row = i / Kd;
  int b = (int)(row % 16);
  int l = (int)(row / 16);
  unsigned short v = 0;
  if (kc < 5 * C) {
    int k = kc / C;
    int c = kc - k * C;
    int ls = l + k - 2;
    if (ls >= 0 && ls < L) v = f2bf(src[((size_t)ls * 16 + b) * C + c]);
  }
  dst[i] = v;
}
__global__ void k_final(const float* __restrict__ mel,
                        const float* __restrict__ c3, float* __restrict__ out) {
  size_t i = (size_t)blockIdx.x * blockDim.x + threadIdx.x;
  size_t total = (size_t)8 * 256 * 80;
  if (i >= total) return;
  int m = (int)(i % 80);
  size_t r = i / 80;
  int t = (int)(r % 256);
  int b = (int)(r / 256);
  size_t src = ((size_t)t * 16 + b) * 80 + m;
  out[i] = mel[src] + c3[src];
}

// ------------------------------- host side ---------------------------------
static inline void launch_gemm(hipStream_t s, const unsigned short* A,
                               const unsigned short* Bw, float* C,
                               const float* b1, const float* b2, int M, int N,
                               int K) {
  int Npad = (N + 63) & ~63;
  int nMac = Npad / 64;
  int total = (M / 16) * nMac;
  int blocks = (total + 7) / 8;
  gemm_bf16_kernel<<<blocks, 256, 0, s>>>(A, Bw, C, b1, b2, M, N, Npad, K);
}

#define EW(n) dim3((unsigned)(((size_t)(n) + 255) / 256)), dim3(256), 0, stream

extern "C" void kernel_launch(void* const* d_in, const int* in_sizes, int n_in,
                              void* d_out, int out_size, void* d_ws,
                              size_t ws_size, hipStream_t stream) {
  (void)in_sizes; (void)n_in; (void)out_size; (void)ws_size;
  // ---- inputs (setup_inputs order, nested lists flattened) ----
  const int* text_tokens = (const int*)d_in[0];
  const float* spk_emb = (const float*)d_in[1];
  const float* mel_target = (const float*)d_in[2];
  const float* emb = (const float*)d_in[4];
  const float *eWih[6], *eWhh[6], *eBih[6], *eBhh[6];
  for (int g = 0; g < 6; ++g) {
    eWih[g] = (const float*)d_in[5 + 4 * g + 0];
    eWhh[g] = (const float*)d_in[5 + 4 * g + 1];
    eBih[g] = (const float*)d_in[5 + 4 * g + 2];
    eBhh[g] = (const float*)d_in[5 + 4 * g + 3];
  }
  const float* Wq = (const float*)d_in[29];
  const float* bq = (const float*)d_in[30];
  const float* Wk = (const float*)d_in[31];
  const float* bk = (const float*)d_in[32];
  const float* Wv = (const float*)d_in[33];
  const float* bv = (const float*)d_in[34];
  const float* Wo = (const float*)d_in[35];
  const float* bo = (const float*)d_in[36];
  const float* spk_W = (const float*)d_in[37];
  const float* spk_b = (const float*)d_in[38];
  const float* dWih0 = (const float*)d_in[39];
  const float* dWhh0 = (const float*)d_in[40];
  const float* dBih0 = (const float*)d_in[41];
  const float* dBhh0 = (const float*)d_in[42];
  const float* dWih1 = (const float*)d_in[43];
  const float* dWhh1 = (const float*)d_in[44];
  const float* dBih1 = (const float*)d_in[45];
  const float* dBhh1 = (const float*)d_in[46];
  const float* melpW = (const float*)d_in[47];
  const float* melpb = (const float*)d_in[48];
  const float* W1 = (const float*)d_in[49];
  const float* b1 = (const float*)d_in[50];
  const float* g1 = (const float*)d_in[51];
  const float* be1 = (const float*)d_in[52];
  const float* m1 = (const float*)d_in[53];
  const float* v1 = (const float*)d_in[54];
  const float* W2 = (const float*)d_in[55];
  const float* b2 = (const float*)d_in[56];
  const float* g2 = (const float*)d_in[57];
  const float* be2 = (const float*)d_in[58];
  const float* m2 = (const float*)d_in[59];
  const float* v2 = (const float*)d_in[60];
  const float* W3 = (const float*)d_in[61];
  const float* b3 = (const float*)d_in[62];

  // ---- workspace carve-up ----
  size_t off = 0;
  auto alloc = [&](size_t bytes) -> void* {
    off = (off + 255) & ~(size_t)255;
    void* p = (char*)d_ws + off;
    off += bytes;
    return p;
  };
  const int encK[6] = {512, 512, 1024, 1024, 1024, 1024};
  unsigned short* encWihB[6];
  unsigned short* encWhhB[6];
  for (int g = 0; g < 6; ++g) {
    encWihB[g] = (unsigned short*)alloc((size_t)2048 * encK[g] * 2);
    encWhhB[g] = (unsigned short*)alloc((size_t)2048 * 512 * 2);
  }
  unsigned short* dWih0B = (unsigned short*)alloc((size_t)4096 * 1120 * 2);
  unsigned short* dWhh0B = (unsigned short*)alloc((size_t)4096 * 1024 * 2);
  unsigned short* dWih1B = (unsigned short*)alloc((size_t)4096 * 1024 * 2);
  unsigned short* dWhh1B = (unsigned short*)alloc((size_t)4096 * 1024 * 2);
  unsigned short* WqB = (unsigned short*)alloc((size_t)1024 * 1024 * 2);
  unsigned short* WkB = (unsigned short*)alloc((size_t)1024 * 1024 * 2);
  unsigned short* WvB = (unsigned short*)alloc((size_t)1024 * 1024 * 2);
  unsigned short* WoB = (unsigned short*)alloc((size_t)1024 * 1024 * 2);
  unsigned short* spkWB = (unsigned short*)alloc((size_t)1024 * 256 * 2);
  unsigned short* melpB = (unsigned short*)alloc((size_t)128 * 1024 * 2);  // 80->128 rows
  unsigned short* W1r = (unsigned short*)alloc((size_t)512 * 416 * 2);
  unsigned short* W2r = (unsigned short*)alloc((size_t)512 * 2560 * 2);
  unsigned short* W3r = (unsigned short*)alloc((size_t)128 * 2560 * 2);      // 80->128 rows

  unsigned short* embX = (unsigned short*)alloc((size_t)2048 * 512 * 2);
  float* GpreF = (float*)alloc((size_t)2048 * 2048 * 4);
  float* GpreBw = (float*)alloc((size_t)2048 * 2048 * 4);
  float* encYa = (float*)alloc((size_t)2048 * 1024 * 4);
  float* encYb = (float*)alloc((size_t)2048 * 1024 * 4);
  unsigned short* encYaB = (unsigned short*)alloc((size_t)2048 * 1024 * 2);
  unsigned short* encYbB = (unsigned short*)alloc((size_t)2048 * 1024 * 2);
  unsigned short* spkA = (unsigned short*)alloc((size_t)16 * 256 * 2);
  float* spkP = (float*)alloc((size_t)16 * 1024 * 4);
  float* encS = (float*)alloc((size_t)2048 * 1024 * 4);
  unsigned short* encSB = (unsigned short*)alloc((size_t)2048 * 1024 * 2);
  unsigned short* qpadB = (unsigned short*)alloc((size_t)4096 * 1024 * 2);
  float* Qf = (float*)alloc((size_t)4096 * 1024 * 4);
  unsigned short* QB = (unsigned short*)alloc((size_t)4096 * 1024 * 2);
  float* Kf = (float*)alloc((size_t)2048 * 1024 * 4);
  unsigned short* KB = (unsigned short*)alloc((size_t)2048 * 1024 * 2);
  float* Vf = (float*)alloc((size_t)2048 * 1024 * 4);
  unsigned short* VtB = (unsigned short*)alloc((size_t)524288 * 2);
  float* attnH = (float*)alloc((size_t)4096 * 1024 * 4);
  unsigned short* attnHB = (unsigned short*)alloc((size_t)4096 * 1024 * 2);
  float* attnO = (float*)alloc((size_t)4096 * 1024 * 4);
  unsigned short* XB = (unsigned short*)alloc((size_t)4096 * 1120 * 2);
  float* GpreD = (float*)alloc((size_t)4096 * 4096 * 4);
  float* h0 = (float*)alloc((size_t)4096 * 1024 * 4);
  unsigned short* h0B = (unsigned short*)alloc((size_t)4096 * 1024 * 2);
  float* h1 = (float*)alloc((size_t)4096 * 1024 * 4);
  unsigned short* h1B = (unsigned short*)alloc((size_t)4096 * 1024 * 2);
  float* melF = (float*)alloc((size_t)4096 * 80 * 4);
  unsigned short* colB = (unsigned short*)alloc((size_t)4096 * 2560 * 2);
  float* convY = (float*)alloc((size_t)4096 * 512 * 4);
  float* bnY = (float*)alloc((size_t)4096 * 512 * 4);
  float* c3 = (float*)alloc((size_t)4096 * 80 * 4);

  // ---- weight conversion / repack to bf16 (Npad,K) row-major ----
  for (int g = 0; g < 6; ++g) {
    k_convert_pad<<<EW((size_t)2048 * encK[g])>>>(eWih[g], encWihB[g], 2048,
                                                  2048, encK[g], encK[g]);
    k_convert_pad<<<EW((size_t)2048 * 512)>>>(eWhh[g], encWhhB[g], 2048, 2048,
                                              512, 512);
  }
  k_convert_pad<<<EW((size_t)4096 * 1120)>>>(dWih0, dWih0B, 4096, 4096, 1104,
                                             1120);
  k_convert_pad<<<EW((size_t)4096 * 1024)>>>(dWhh0, dWhh0B, 4096, 4096, 1024,
                                             1024);
  k_convert_pad<<<EW((size_t)4096 * 1024)>>>(dWih1, dWih1B, 4096, 4096, 1024,
                                             1024);
  k_convert_pad<<<EW((size_t)4096 * 1024)>>>(dWhh1, dWhh1B, 4096, 4096, 1024,
                                             1024);
  k_convert_pad<<<EW((size_t)1024 * 1024)>>>(Wq, WqB, 1024, 1024, 1024, 1024);
  k_convert_pad<<<EW((size_t)1024 * 1024)>>>(Wk, WkB, 1024, 1024, 1024, 1024);
  k_convert_pad<<<EW((size_t)1024 * 1024)>>>(Wv, WvB, 1024, 1024, 1024, 1024);
  k_convert_pad<<<EW((size_t)1024 * 1024)>>>(Wo, WoB, 1024, 1024, 1024, 1024);
  k_convert_pad<<<EW((size_t)1024 * 256)>>>(spk_W, spkWB, 1024, 1024, 256, 256);
  k_convert_pad<<<EW((size_t)128 * 1024)>>>(melpW, melpB, 80, 128, 1024, 1024);
  k_conv_repack<<<EW((size_t)512 * 416)>>>(W1, W1r, 512, 512, 80, 416);
  k_conv_repack<<<EW((size_t)512 * 2560)>>>(W2, W2r, 512, 512, 512, 2560);
  k_conv_repack<<<EW((size_t)128 * 2560)>>>(W3, W3r, 80, 128, 512, 2560);

  // ---- encoder: 3-layer BiLSTM ----
  k_embed<<<EW((size_t)2048 * 512)>>>(text_tokens, emb, embX);
  const unsigned short* encIn = embX;
  int Kcur = 512;
  float* outs[3] = {encYa, encYb, encYa};
  unsigned short* outsB[3] = {encYaB, encYbB, encYaB};
  for (int l = 0; l < 3; ++l) {
    launch_gemm(stream, encIn, encWihB[2 * l], GpreF, eBih[2 * l],
                eBhh[2 * l], 2048, 2048, Kcur);
    launch_gemm(stream, encIn, encWihB[2 * l + 1], GpreBw, eBih[2 * l + 1],
                eBhh[2 * l + 1], 2048, 2048, Kcur);
    lstm_scan_kernel<512, 1><<<2, 1024, 0, stream>>>(
        GpreF, GpreBw, encWhhB[2 * l], encWhhB[2 * l + 1], outs[l], 1024, 0,
        512, S_);
    k_convert<<<EW((size_t)2048 * 1024)>>>(outs[l], outsB[l],
                                           (size_t)2048 * 1024);
    encIn = outsB[l];
    Kcur = 1024;
  }
  // ---- speaker projection + add ----
  k_spkA<<<EW(16 * 256)>>>(spk_emb, spkA);
  launch_gemm(stream, spkA, spkWB, spkP, spk_b, nullptr, 16, 1024, 256);
  k_spk_add<<<EW((size_t)2048 * 1024)>>>(outs[2], spkP, encS, encSB);

  // ---- MHA ----
  k_qpad<<<EW((size_t)4096 * 1024)>>>(mel_target, qpadB);
  launch_gemm(stream, qpadB, WqB, Qf, bq, nullptr, 4096, 1024, 1024);
  k_convert<<<EW((size_t)4096 * 1024)>>>(Qf, QB, (size_t)4096 * 1024);
  launch_gemm(stream, encSB, WkB, Kf, bk, nullptr, 2048, 1024, 1024);
  k_convert<<<EW((size_t)2048 * 1024)>>>(Kf, KB, (size_t)2048 * 1024);
  launch_gemm(stream, encSB, WvB, Vf, bv, nullptr, 2048, 1024, 1024);
  k_vt<<<EW((size_t)524288)>>>(Vf, VtB);
  attention_kernel<<<256, 256, 0, stream>>>(QB, KB, VtB, attnH);
  k_convert<<<EW((size_t)4096 * 1024)>>>(attnH, attnHB, (size_t)4096 * 1024);
  launch_gemm(stream, attnHB, WoB, attnO, bo, nullptr, 4096, 1024, 1024);

  // ---- decoder LSTMs ----
  k_buildX<<<EW((size_t)4096 * 1120)>>>(attnO, mel_target, XB);
  launch_gemm(stream, XB, dWih0B, GpreD, dBih0, dBhh0, 4096, 4096, 1120);
  lstm_scan_kernel<1024, 2><<<1, 1024, 0, stream>>>(
      GpreD, GpreD, dWhh0B, dWhh0B, h0, 1024, 0, 0, T_);
  k_convert<<<EW((size_t)4096 * 1024)>>>(h0, h0B, (size_t)4096 * 1024);
  launch_gemm(stream, h0B, dWih1B, GpreD, dBih1, dBhh1, 4096, 4096, 1024);
  lstm_scan_kernel<1024, 2><<<1, 1024, 0, stream>>>(
      GpreD, GpreD, dWhh1B, dWhh1B, h1, 1024, 0, 0, T_);
  k_convert<<<EW((size_t)4096 * 1024)>>>(h1, h1B, (size_t)4096 * 1024);
  launch_gemm(stream, h1B, melpB, melF, melpb, nullptr, 4096, 80, 1024);

  // ---- postnet: im2col + WMMA GEMM convs ----
  k_im2col<<<EW((size_t)4096 * 416)>>>(melF, colB, 256, 80, 416);
  launch_gemm(stream, colB, W1r, convY, b1, nullptr, 4096, 512, 416);
  k_bn_tanh<<<EW((size_t)4096 * 512)>>>(convY, g1, be1, m1, v1, bnY, 512,
                                        (size_t)4096 * 512);
  k_im2col<<<EW((size_t)4096 * 2560)>>>(bnY, colB, 256, 512, 2560);
  launch_gemm(stream, colB, W2r, convY, b2, nullptr, 4096, 512, 2560);
  k_bn_tanh<<<EW((size_t)4096 * 512)>>>(convY, g2, be2, m2, v2, bnY, 512,
                                        (size_t)4096 * 512);
  k_im2col<<<EW((size_t)4096 * 2560)>>>(bnY, colB, 256, 512, 2560);
  launch_gemm(stream, colB, W3r, c3, b3, nullptr, 4096, 80, 2560);

  k_final<<<EW((size_t)8 * 256 * 80)>>>(melF, c3, (float*)d_out);
}
